// Difference_RNN_81166291960583
// MI455X (gfx1250) — compile-verified, matrix operand order checked by reading, not run on hardware
//
#include <hip/hip_runtime.h>
#include <hip/hip_bf16.h>

// ---------------------------------------------------------------------------
// Difference RNN: x_t = W_A x_{t-1} + (W_B u_t + b_A + b_B),  D=4096, C=512, T=2048
// Chunked linear scan: S=32-step chunks, NC=64 chunks batched as GEMM columns.
// All heavy math: v_wmma_f32_16x16x32_bf16 (fp32 accumulate).
// All WMMA operand loads are contiguous 32B/lane (A row-major, B col-major)
// and are kept in the global address space (global_load_b128, no FLAT).
// The power chain keeps (row-major, col-major) bf16 copies of each W_A^2^i,
// produced directly from the WMMA D-fragment layout (no separate transposes).
// ---------------------------------------------------------------------------

typedef __bf16 bf16_t;
typedef __attribute__((ext_vector_type(16))) __bf16 v16bf;
typedef __attribute__((ext_vector_type(8)))  __bf16 v8bf;
typedef __attribute__((ext_vector_type(8)))  float  v8f;

#define D_DIM 4096
#define C_DIM 512
#define T_DIM 2048
#define S_CH  32
#define NC_CH 64   // T / S

// ---------------- elementwise helpers ----------------

__global__ void k_f32_to_bf16(const float* __restrict__ in, bf16_t* __restrict__ out, int n) {
    int i = blockIdx.x * blockDim.x + threadIdx.x;
    if (i < n) out[i] = (bf16_t)in[i];
}

__global__ void k_bias_sum(const float* __restrict__ a, const float* __restrict__ b,
                           float* __restrict__ o, int n) {
    int i = blockIdx.x * blockDim.x + threadIdx.x;
    if (i < n) o[i] = a[i] + b[i];
}

// LDS-tiled transpose + fp32->bf16 convert: out[c*R + r] = (bf16) in[r*Cc + c].
// Grid: (Cc/32, R/32), block (32, 8). Coalesced on both sides.
__global__ void k_transpose_to_bf16(const float* __restrict__ in, bf16_t* __restrict__ out,
                                    int R, int Cc) {
    __shared__ float tile[32][33];
    const int c0 = blockIdx.x * 32, r0 = blockIdx.y * 32;
#pragma unroll
    for (int j = 0; j < 4; ++j) {
        int r = r0 + threadIdx.y + j * 8;
        tile[threadIdx.y + j * 8][threadIdx.x] = in[(size_t)r * Cc + c0 + threadIdx.x];
    }
    __syncthreads();
#pragma unroll
    for (int j = 0; j < 4; ++j) {
        int c = c0 + threadIdx.y + j * 8;
        out[(size_t)c * R + r0 + threadIdx.x] = (bf16_t)tile[threadIdx.x][threadIdx.y + j * 8];
    }
}

// ---------------- unified WMMA GEMM -----------------------------------------
// A row-major [M][K] (lda = K), B col-major per column ([n][k], ldb = K).
// Wave computes a 16 x (NT*16) tile; block = (blockDim/32) waves stacked in M.
// Grid: x = N/(NT*16), y = M/(waves*16).
//   D[m][n] = sum_k A[m][k]*B[n][k]  (+ Cadd[n*cstride + m])
// Outputs (each optional):
//   O      fp32:            O[n*ostride + m]
//   ObfCol bf16 col-major:  ObfCol[n*obld + m]   (contiguous 16B/lane)
//   ObfRow bf16 row-major:  ObfRow[m*orld + n]   (feeds next A operand)

template <int NT>
__global__ void k_wmma_gemm(const bf16_t* __restrict__ A, int lda,
                            const bf16_t* __restrict__ B, int ldb,
                            const float* __restrict__ Cadd, size_t cstride,
                            float* __restrict__ O, size_t ostride,
                            bf16_t* __restrict__ ObfCol, int obld,
                            bf16_t* __restrict__ ObfRow, int orld,
                            int K) {
    const int lane  = threadIdx.x & 31;
    const int wave  = threadIdx.x >> 5;
    const int waves = blockDim.x >> 5;
    const int lr    = lane & 15;
    const int half  = lane >> 4;
    const int tm    = blockIdx.y * (waves * 16) + wave * 16;
    const int tn    = blockIdx.x * (NT * 16);

    // Direct indexing off the __restrict__ kernel args (no pointer arrays) so
    // address-space inference keeps every load a GLOBAL (not FLAT) access.
    const size_t arow  = (size_t)(tm + lr) * lda + half * 16;
    const size_t brow  = (size_t)(tn + lr) * ldb + half * 16;
    const size_t bstep = (size_t)16 * ldb;           // between N sub-tiles

    v8f acc[NT];
#pragma unroll
    for (int t = 0; t < NT; ++t) acc[t] = (v8f){};

    for (int k = 0; k < K; k += 32) {
        v16bf a = *(const v16bf*)(A + arow + k);
#pragma unroll
        for (int t = 0; t < NT; ++t) {
            v16bf b = *(const v16bf*)(B + brow + (size_t)t * bstep + k);
            acc[t] = __builtin_amdgcn_wmma_f32_16x16x32_bf16(false, a, false, b,
                                                             (short)0, acc[t], false, false);
        }
    }

    const int mb = tm + half * 8;   // 8 consecutive M values per lane (C/D layout)
#pragma unroll
    for (int t = 0; t < NT; ++t) {
        v8f c = acc[t];
        const int n = tn + t * 16 + lr;
        if (Cadd) c = c + *(const v8f*)(Cadd + (size_t)n * cstride + mb);
        if (O) *(v8f*)(O + (size_t)n * ostride + mb) = c;
        if (ObfCol) {
            v8bf h;
#pragma unroll
            for (int r = 0; r < 8; ++r) h[r] = (bf16_t)c[r];
            *(v8bf*)(ObfCol + (size_t)n * obld + mb) = h;
        }
        if (ObfRow) {
#pragma unroll
            for (int r = 0; r < 8; ++r)
                ObfRow[(size_t)(mb + r) * orld + n] = (bf16_t)c[r];
        }
    }
}

// ---------------- bf16 matvec (stage B carry chain): y = A*x + add ----------
// One wave per row; 256-thread block handles 8 rows. Coalesced 64B/wave/iter.

__global__ void k_matvec_bf16(const bf16_t* __restrict__ A, const float* __restrict__ x,
                              const float* __restrict__ add, float* __restrict__ y, int n) {
    const int lane = threadIdx.x & 31;
    const int wave = threadIdx.x >> 5;
    const int row  = blockIdx.x * 8 + wave;
    if (row >= n) return;
    const size_t rbase = (size_t)row * n;
    float s = 0.f;
    for (int k = lane; k < n; k += 32) s += (float)A[rbase + k] * x[k];
#pragma unroll
    for (int off = 16; off > 0; off >>= 1) s += __shfl_xor(s, off, 32);
    if (lane == 0) y[row] = s + add[row];
}

// ---------------------------------------------------------------------------

extern "C" void kernel_launch(void* const* d_in, const int* in_sizes, int n_in,
                              void* d_out, int out_size, void* d_ws, size_t ws_size,
                              hipStream_t stream) {
    (void)in_sizes; (void)n_in; (void)out_size; (void)ws_size;

    const float* x0 = (const float*)d_in[0];
    const float* u  = (const float*)d_in[1];   // [C][T]
    const float* WA = (const float*)d_in[2];   // [D][D]
    const float* bA = (const float*)d_in[3];
    const float* WB = (const float*)d_in[4];   // [D][C]
    const float* bB = (const float*)d_in[5];
    float* out = (float*)d_out;                // [T][D]

    const int D = D_DIM, C = C_DIM, T = T_DIM, S = S_CH, NC = NC_CH;

    char* w = (char*)d_ws;
    auto alloc = [&](size_t bytes) -> char* {
        char* p = w; w += (bytes + 255) & ~(size_t)255; return p;
    };
    bf16_t* WAbf  = (bf16_t*)alloc((size_t)D * D * 2);   // W_A bf16, row-major
    bf16_t* WAcol = (bf16_t*)alloc((size_t)D * D * 2);   // W_A bf16, col-major ([n][k]); reused as PC1
    bf16_t* WBbf  = (bf16_t*)alloc((size_t)D * C * 2);   // W_B bf16, row-major
    bf16_t* Utbf  = (bf16_t*)alloc((size_t)T * C * 2);   // U^T bf16 ([t][c], B operand for phase 1)
    float*  bsum  = (float*) alloc((size_t)D * 4);       // b_A + b_B
    float*  V     = (float*) alloc((size_t)T * D * 4);   // v_t = W_B u_t + bsum, [T][D]
    bf16_t* PR0   = (bf16_t*)alloc((size_t)D * D * 2);   // power chain, row-major ping
    bf16_t* PC0   = (bf16_t*)alloc((size_t)D * D * 2);   // power chain, col-major ping
    bf16_t* PR1   = (bf16_t*)alloc((size_t)D * D * 2);   // power chain, row-major pong
    bf16_t* PC1   = WAcol;                               // power chain, col-major pong (reuse)
    bf16_t* Y0    = (bf16_t*)alloc((size_t)NC * D * 2);  // stage-A state, col-major [NC][D]
    bf16_t* Y1    = (bf16_t*)alloc((size_t)NC * D * 2);
    float*  YAf   = (float*) alloc((size_t)NC * D * 4);  // stage-A final columns (fp32)
    float*  CF    = (float*) alloc((size_t)NC * D * 4);  // chunk-boundary carries (fp32)
    bf16_t* Z0    = (bf16_t*)alloc((size_t)NC * D * 2);  // stage-C state ping
    bf16_t* Z1    = (bf16_t*)alloc((size_t)NC * D * 2);

    // ---- 0) precision conversion / transposes / bias fold ------------------
    k_f32_to_bf16<<<(D * D + 255) / 256, 256, 0, stream>>>(WA, WAbf, D * D);
    k_f32_to_bf16<<<(D * C + 255) / 256, 256, 0, stream>>>(WB, WBbf, D * C);
    k_bias_sum<<<(D + 255) / 256, 256, 0, stream>>>(bA, bB, bsum, D);
    {   // WAcol[n*D + k] = W_A[k][n]
        dim3 g(D / 32, D / 32), b(32, 8);
        k_transpose_to_bf16<<<g, b, 0, stream>>>(WA, WAcol, D, D);
    }
    {   // Utbf[t*C + c] = U[c][t]
        dim3 g(T / 32, C / 32), b(32, 8);
        k_transpose_to_bf16<<<g, b, 0, stream>>>(u, Utbf, C, T);
    }

    // ---- 1) V[t][d] = sum_c W_B[d][c] U[c][t] + bsum[d]  (M=D,N=T,K=C) -----
    {
        dim3 g(T / 32, D / 64);
        k_wmma_gemm<2><<<g, 128, 0, stream>>>(WBbf, C, Utbf, C,
                                              bsum, (size_t)0,
                                              V, (size_t)D,
                                              nullptr, 0, nullptr, 0, C);
    }

    // ---- 2) W_A^32 via repeated squaring -----------------------------------
    // Each squaring emits both row-major (next A operand) and col-major
    // (next B operand) bf16 copies straight from the D-fragment.
    {
        dim3 g(D / 64, D / 128);   // NT=4, 256 threads: 128x64 block tile
        k_wmma_gemm<4><<<g, 256, 0, stream>>>(WAbf, D, WAcol, D, nullptr, 0, nullptr, 0,
                                              PC0, D, PR0, D, D);  // A^2
        k_wmma_gemm<4><<<g, 256, 0, stream>>>(PR0,  D, PC0,  D, nullptr, 0, nullptr, 0,
                                              PC1, D, PR1, D, D);  // A^4
        k_wmma_gemm<4><<<g, 256, 0, stream>>>(PR1,  D, PC1,  D, nullptr, 0, nullptr, 0,
                                              PC0, D, PR0, D, D);  // A^8
        k_wmma_gemm<4><<<g, 256, 0, stream>>>(PR0,  D, PC0,  D, nullptr, 0, nullptr, 0,
                                              PC1, D, PR1, D, D);  // A^16
        k_wmma_gemm<4><<<g, 256, 0, stream>>>(PR1,  D, PC1,  D, nullptr, 0, nullptr, 0,
                                              nullptr, 0, PR0, D, D);  // A^32 (row only)
    }

    // ---- 3) Stage A: zero-init local scans, 64 chunks batched (N=NC) -------
    //        y_j = W_A y_{j-1} + v[k*S + j],   j = 1..S
    hipMemsetAsync(Y0, 0, (size_t)NC * D * 2, stream);
    {
        dim3 g(NC / 32, D / 64);   // 128 blocks
        for (int j = 1; j <= S; ++j) {
            bf16_t* bin  = ((j - 1) & 1) ? Y1 : Y0;
            bf16_t* bout = (j & 1) ? Y1 : Y0;
            k_wmma_gemm<2><<<g, 128, 0, stream>>>(WAbf, D, bin, D,
                                                  V + (size_t)(j - 1) * D, (size_t)S * D,
                                                  YAf, (size_t)D,
                                                  bout, D, nullptr, 0, D);
        }
    }

    // ---- 4) Stage B: carry chain across chunk boundaries -------------------
    //        b_0 = x_0;  b_{k+1} = A^32 b_k + yA_k
    hipMemcpyAsync(CF, x0, (size_t)D * 4, hipMemcpyDeviceToDevice, stream);
    for (int k = 0; k < NC - 1; ++k) {
        k_matvec_bf16<<<D / 8, 256, 0, stream>>>(PR0, CF + (size_t)k * D,
                                                 YAf + (size_t)k * D,
                                                 CF + (size_t)(k + 1) * D, D);
    }
    k_f32_to_bf16<<<(NC * D + 255) / 256, 256, 0, stream>>>(CF, Z0, NC * D);

    // ---- 5) Stage C: correct-init scans, write x_t to out ------------------
    //        z_j = W_A z_{j-1} + v[k*S + j];  out[k*S + j - 1] = z_j
    {
        dim3 g(NC / 32, D / 64);
        for (int j = 1; j <= S; ++j) {
            bf16_t* bin  = ((j - 1) & 1) ? Z1 : Z0;
            bf16_t* bout = (j & 1) ? Z1 : Z0;
            k_wmma_gemm<2><<<g, 128, 0, stream>>>(WAbf, D, bin, D,
                                                  V + (size_t)(j - 1) * D, (size_t)S * D,
                                                  out + (size_t)(j - 1) * D, (size_t)S * D,
                                                  bout, D, nullptr, 0, D);
        }
    }
}
